// MultiheadAttentionQ_52664888983787
// MI455X (gfx1250) — compile-verified, matrix operand order checked by reading, not run on hardware
//
#include <hip/hip_runtime.h>
#include <hip/hip_bf16.h>

// ---------------------------------------------------------------------------
// Quantized multi-head attention forward for MI455X (gfx1250, wave32, WMMA).
//
// Data movement: GLOBAL_LOAD_ASYNC_TO_LDS_B128 (ASYNCcnt) double-buffered
// B-panel / K/V staging shared by the 8 waves of a block; branch-free
// two-step loop bodies so accumulators keep fixed registers (no mov/nop
// stitching); grouped ds_load operand fetch so one s_wait_dscnt feeds a
// burst of 4 v_wmma_f32_16x16x32_f16.
// ---------------------------------------------------------------------------

typedef _Float16 half_t;
typedef __attribute__((ext_vector_type(16))) _Float16 v16h;
typedef __attribute__((ext_vector_type(8)))  _Float16 v8h;
typedef __attribute__((ext_vector_type(8)))  float    v8f;

#define S_LEN   2048
#define BATCH   2
#define D_EMB   1024
#define N3      3072
#define HEADS   16
#define HD      64
#define BH      32          // BATCH*HEADS
#define MROWS   4096        // S_LEN*BATCH

// ---- helpers --------------------------------------------------------------

// Load one lane's 16 halves of a 16x32 f16 WMMA A/B operand.
// ISA layout: lane L holds row (L&15); its K indices are two contiguous
// 8-element chunks at k0 = 8*(L>>4) and k0+16.
__device__ __forceinline__ v16h load_ab16(const half_t* p) {
  v8h lo = *(const v8h*)p;
  v8h hi = *(const v8h*)(p + 16);
  return __builtin_shufflevector(lo, hi, 0,1,2,3,4,5,6,7,8,9,10,11,12,13,14,15);
}

__device__ __forceinline__ v8f wmma_f16(v16h a, v16h b, v8f c) {
  // D = A(16x32 f16) * B(32x16 f16) + C(16x16 f32)
  return __builtin_amdgcn_wmma_f32_16x16x32_f16(false, a, false, b,
                                                (short)0, c, false, false);
}

// Async global -> LDS copy, 16 B per lane, tracked with ASYNCcnt.
// Generic LDS pointers carry the LDS offset in their low 32 bits
__device__ __forceinline__ void async_copy_b128(void* lds, const void* gsrc) {
  unsigned           l = (unsigned)(size_t)lds;
  unsigned long long g = (unsigned long long)(size_t)gsrc;
  asm volatile("global_load_async_to_lds_b128 %0, %1, off"
               :: "v"(l), "v"(g) : "memory");
}
__device__ __forceinline__ void wait_async0() {
  asm volatile("s_wait_asynccnt 0x0" ::: "memory");
}
__device__ __forceinline__ void wait_async1() {
  asm volatile("s_wait_asynccnt 0x1" ::: "memory");
}
__device__ __forceinline__ void wait_async2() {
  asm volatile("s_wait_asynccnt 0x2" ::: "memory");
}

__device__ __forceinline__ float wave_max32(float v) {
#pragma unroll
  for (int m = 16; m >= 1; m >>= 1) v = fmaxf(v, __shfl_xor(v, m, 32));
  return v;
}

__device__ __forceinline__ float scale_from(const unsigned int* slot) {
  float am = __uint_as_float(*slot);           // non-negative absmax
  return fmaxf(am * (1.0f / 127.0f), 1e-8f);
}

__device__ __forceinline__ float fq_val(float x, float s) {
  float q = rintf(x / s);                      // jnp.round == RNE
  q = fminf(fmaxf(q, -128.0f), 127.0f);
  return q * s;
}

// ---- small elementwise / reduction kernels --------------------------------

__global__ void init_scalars_k(unsigned int* slots) {
  if (threadIdx.x < 16) slots[threadIdx.x] = 0u;
}

__global__ void absmax_f32_k(const float* __restrict__ src, int n,
                             unsigned int* __restrict__ slot) {
  float m = 0.0f;
  for (int i = blockIdx.x * blockDim.x + threadIdx.x; i < n;
       i += gridDim.x * blockDim.x)
    m = fmaxf(m, fabsf(src[i]));
  m = wave_max32(m);
  if ((threadIdx.x & 31) == 0) atomicMax(slot, __float_as_uint(m));
}

__global__ void cvt_f32_f16_k(const float* __restrict__ src,
                              half_t* __restrict__ dst, int n) {
  int i = blockIdx.x * blockDim.x + threadIdx.x;
  if (i < n) dst[i] = (half_t)src[i];
}

// fake_quant f32 -> f16 (used for weights and for `heads`)
__global__ void fq_f32_to_f16_k(const float* __restrict__ src,
                                half_t* __restrict__ dst, int n,
                                const unsigned int* __restrict__ slot) {
  int i = blockIdx.x * blockDim.x + threadIdx.x;
  if (i >= n) return;
  float s = scale_from(slot);
  dst[i] = (half_t)fq_val(src[i], s);
}

// fake_quant f16 buffer in place: y = fq(x * premul, scale(slotA));
// optionally also reduce absmax(|y| * bmul) into slotB.
__global__ void fq_f16_inplace_k(half_t* __restrict__ buf, int n,
                                 const unsigned int* __restrict__ slotA,
                                 float premul,
                                 unsigned int* __restrict__ slotB,
                                 float bmul) {
  int i = blockIdx.x * blockDim.x + threadIdx.x;
  float s = scale_from(slotA);
  float lm = 0.0f;
  if (i < n) {
    float y = fq_val((float)buf[i] * premul, s);
    buf[i] = (half_t)y;
    lm = fabsf(y) * bmul;
  }
  if (slotB) {
    lm = wave_max32(lm);
    if ((threadIdx.x & 31) == 0) atomicMax(slotB, __float_as_uint(lm));
  }
}

__global__ void fq_f32_inplace_k(float* __restrict__ buf, int n,
                                 const unsigned int* __restrict__ slot) {
  int i = blockIdx.x * blockDim.x + threadIdx.x;
  if (i >= n) return;
  float s = scale_from(slot);
  buf[i] = fq_val(buf[i], s);
}

// ---- shared GEMM K-step: grouped B ds_loads, then 4 back-to-back WMMAs ----
__device__ __forceinline__ void gemm_step(const half_t* __restrict__ bbuf,
                                          v16h a, v8f acc[4], int nl, int kc) {
  v16h b0 = load_ab16(&bbuf[(0 * 16 + nl) * 32 + kc]);
  v16h b1 = load_ab16(&bbuf[(1 * 16 + nl) * 32 + kc]);
  v16h b2 = load_ab16(&bbuf[(2 * 16 + nl) * 32 + kc]);
  v16h b3 = load_ab16(&bbuf[(3 * 16 + nl) * 32 + kc]);
  acc[0] = wmma_f16(a, b0, acc[0]);
  acc[1] = wmma_f16(a, b1, acc[1]);
  acc[2] = wmma_f16(a, b2, acc[2]);
  acc[3] = wmma_f16(a, b3, acc[3]);
}

// ---- input-projection GEMM ------------------------------------------------
// C = X(4096x1024) @ W^T (W = 3072x1024 row-major) + bias.
// Block = 8 waves sharing one 64-column B panel; panel K-chunks (64x32 f16,
// 4 KB) double-buffered in LDS via async global->LDS.  Two K-steps per loop
// iteration (branch-free; buffer indices compile-time), A software-pipelined.
// absmax over ALL columns; only columns in [c0, c0+1024) scatter-stored into
// per-head layout: out[bh*S*HD + s*out_ss + d*out_sd], bh=b*16+h, m=s*2+b.
__global__ void gemm_xw_k(const half_t* __restrict__ X,
                          const half_t* __restrict__ W,
                          const float* __restrict__ bias,
                          half_t* __restrict__ outSlice,
                          int c0, int out_ss, int out_sd,
                          unsigned int* __restrict__ slot) {
  __shared__ alignas(16) half_t bsm[2][64 * 32];
  const int tid  = threadIdx.x;
  const int lane = tid & 31;
  const int wib  = tid >> 5;
  const int ng   = blockIdx.x % (N3 / 64);     // B-panel index, shared by block
  const int mtg  = blockIdx.x / (N3 / 64);
  const int mt   = mtg * 8 + wib;              // 16-row tile per wave
  const int nl   = lane & 15;
  const int kc   = (lane >> 4) << 3;           // 0 or 8
  const int ngb  = ng * 64;
  const int srow = tid >> 2;
  const int soff = srow * 32 + (tid & 3) * 8;  // staging LDS offset (halves)

  v8f acc[4] = {v8f{}, v8f{}, v8f{}, v8f{}};
  const half_t* xrow = X + (size_t)(mt * 16 + nl) * D_EMB + kc;
  const half_t* wst  = W + (size_t)(ngb + srow) * D_EMB + (tid & 3) * 8;

  async_copy_b128(&bsm[0][soff], wst);         // stage K-chunk 0
  v16h a = load_ab16(xrow);

#pragma unroll 1
  for (int kk = 0; kk < D_EMB - 64; kk += 64) {
    // step A: consume bsm[0] (kk), stage bsm[1] (kk+32)
    async_copy_b128(&bsm[1][soff], wst + kk + 32);
    wait_async1();
    __syncthreads();
    __builtin_prefetch(xrow + kk + 512, 0, 3);     // global_prefetch_b8 (A)
    v16h a1 = load_ab16(xrow + kk + 32);           // pipeline next A
    gemm_step(bsm[0], a, acc, nl, kc);
    __syncthreads();
    // step B: consume bsm[1] (kk+32), stage bsm[0] (kk+64)
    async_copy_b128(&bsm[0][soff], wst + kk + 64);
    wait_async1();
    __syncthreads();
    v16h a2 = load_ab16(xrow + kk + 64);
    gemm_step(bsm[1], a1, acc, nl, kc);
    a = a2;
    __syncthreads();
  }
  // tail: kk = D_EMB-64 and D_EMB-32
  async_copy_b128(&bsm[1][soff], wst + (D_EMB - 32));
  wait_async1();
  __syncthreads();
  v16h a1 = load_ab16(xrow + (D_EMB - 32));
  gemm_step(bsm[0], a, acc, nl, kc);
  __syncthreads();
  wait_async0();
  __syncthreads();
  gemm_step(bsm[1], a1, acc, nl, kc);

  // epilogue: bias, absmax over all columns, slice scatter-store
  float lmax = 0.0f;
#pragma unroll
  for (int t = 0; t < 4; ++t) {
    int nglob = ngb + t * 16 + nl;
    float bv = bias[nglob];
    bool insl = (nglob >= c0) && (nglob < c0 + D_EMB);
    int col = nglob - c0;
    int h = (col >> 6) & 15, d = col & 63;
#pragma unroll
    for (int r = 0; r < 8; ++r) {
      float v = acc[t][r] + bv;            // C layout: m=r+8*(lane>>4), n=nl
      lmax = fmaxf(lmax, fabsf(v));
      if (insl) {
        int m = mt * 16 + r + ((lane >> 4) << 3);
        int s = m >> 1, b = m & 1;
        int bh = b * HEADS + h;
        outSlice[(size_t)bh * (S_LEN * HD) + (size_t)s * out_ss +
                 (size_t)d * out_sd] = (half_t)v;
      }
    }
  }
  lmax = wave_max32(lmax);
  if (lane == 0) atomicMax(slot, __float_as_uint(lmax));
}

// ---- flash attention per-key-block compute --------------------------------
__device__ __forceinline__ void flash_block(
    const half_t* __restrict__ kbuf,   // 32 x 64 f16 (key-major)
    const half_t* __restrict__ vbuf,   // 64 x 32 f16 (d-major)
    half_t* __restrict__ pst,          // per-wave 16x32 staging
    v16h aq0, v16h aq1, v8f o[4], float rm[8], float rs[8],
    int nl, int kc, int mofs) {
  // --- S = Q @ K^T (two 16-key tiles, K=64 -> 2 WMMA each); group loads
  v16h bk00 = load_ab16(&kbuf[(size_t)nl * HD + kc]);
  v16h bk01 = load_ab16(&kbuf[(size_t)nl * HD + kc + 32]);
  v16h bk10 = load_ab16(&kbuf[(size_t)(16 + nl) * HD + kc]);
  v16h bk11 = load_ab16(&kbuf[(size_t)(16 + nl) * HD + kc + 32]);
  v8f s0 = v8f{}, s1 = v8f{};
  s0 = wmma_f16(aq0, bk00, s0);
  s1 = wmma_f16(aq0, bk10, s1);
  s0 = wmma_f16(aq1, bk01, s0);
  s1 = wmma_f16(aq1, bk11, s1);

  // --- online softmax (row stats redundantly in each 16-lane group)
  float alpha[8];
#pragma unroll
  for (int r = 0; r < 8; ++r) {
    float mx = fmaxf(s0[r], s1[r]);
#pragma unroll
    for (int msk = 8; msk >= 1; msk >>= 1) mx = fmaxf(mx, __shfl_xor(mx, msk, 32));
    float newm = fmaxf(rm[r], mx);
    alpha[r] = __expf(rm[r] - newm);
    float p0 = __expf(s0[r] - newm);
    float p1 = __expf(s1[r] - newm);
    float ps = p0 + p1;
#pragma unroll
    for (int msk = 8; msk >= 1; msk >>= 1) ps += __shfl_xor(ps, msk, 32);
    rs[r] = rs[r] * alpha[r] + ps;
    rm[r] = newm;
    int m = r + mofs;
    pst[m * 32 + nl]      = (half_t)p0;   // key cols 0..15
    pst[m * 32 + 16 + nl] = (half_t)p1;   // key cols 16..31
  }
#pragma unroll
  for (int t = 0; t < 4; ++t)
#pragma unroll
    for (int r = 0; r < 8; ++r) o[t][r] *= alpha[r];

  // --- C-layout -> A-layout via per-wave LDS; same-wave LDS is in-order,
  //     fence compiler + DS pipe explicitly (CDNA5 split counters).
  asm volatile("s_wait_dscnt 0" ::: "memory");
  v16h ap = load_ab16(&pst[nl * 32 + kc]);

  // --- O += P(16x32) @ V(32x64); grouped V-operand loads
  v16h bv0 = load_ab16(&vbuf[(size_t)(0 * 16 + nl) * 32 + kc]);
  v16h bv1 = load_ab16(&vbuf[(size_t)(1 * 16 + nl) * 32 + kc]);
  v16h bv2 = load_ab16(&vbuf[(size_t)(2 * 16 + nl) * 32 + kc]);
  v16h bv3 = load_ab16(&vbuf[(size_t)(3 * 16 + nl) * 32 + kc]);
  o[0] = wmma_f16(ap, bv0, o[0]);
  o[1] = wmma_f16(ap, bv1, o[1]);
  o[2] = wmma_f16(ap, bv2, o[2]);
  o[3] = wmma_f16(ap, bv3, o[3]);
}

// ---- flash attention ------------------------------------------------------
// One wave: one (bh, 16-row q-tile).  The 8 waves of a block share the same
// head (consecutive wids => same bh): each 32-key block's K (32x64) and
// V^T (64x32) tiles are async-staged once into double-buffered LDS.
__global__ void flash_attn_k(const half_t* __restrict__ qh,
                             const half_t* __restrict__ kh,
                             const half_t* __restrict__ vT,
                             float* __restrict__ heads,
                             unsigned int* __restrict__ slot) {
  __shared__ alignas(16) half_t ksm[2][32 * HD];     // 2 x 4 KB
  __shared__ alignas(16) half_t vsm[2][HD * 32];     // 2 x 4 KB
  __shared__ alignas(16) half_t pstage[8][16 * 32];  // per-wave P, 8 KB
  const int tid  = threadIdx.x;
  const int lane = tid & 31;
  const int wib  = tid >> 5;
  const int wid  = blockIdx.x * 8 + wib;
  const int bh = wid >> 7;                     // 0..31 (same for whole block)
  const int qt = wid & 127;                    // 0..127
  const int nl = lane & 15;
  const int kc = (lane >> 4) << 3;             // 0 or 8
  const int mofs = (lane >> 4) << 3;           // C/A row group offset
  const int srow = tid >> 2, schk = (tid & 3) * 8;   // V staging row/chunk

  const half_t* qbase = qh + (size_t)(bh * S_LEN + qt * 16) * HD;
  const half_t* kbase = kh + (size_t)bh * S_LEN * HD;   // rows contiguous
  const half_t* vbase = vT + (size_t)bh * HD * S_LEN;   // d-major

  // Q tile (16x64): two A operands covering K 0..31 and 32..63, resident.
  v16h aq0 = load_ab16(qbase + (size_t)nl * HD + kc);
  v16h aq1 = load_ab16(qbase + (size_t)nl * HD + kc + 32);

  v8f o[4] = {v8f{}, v8f{}, v8f{}, v8f{}};
  float rm[8], rs[8];
#pragma unroll
  for (int r = 0; r < 8; ++r) { rm[r] = -INFINITY; rs[r] = 0.0f; }

  // stage key-block 0: K tile is one contiguous 4 KB blob; V tile is 64 rows
  // of 64 B (4 chunks each) from the d-major V^T.
  async_copy_b128(&ksm[0][tid * 8], kbase + (size_t)tid * 8);
  async_copy_b128(&vsm[0][srow * 32 + schk],
                  vbase + (size_t)srow * S_LEN + schk);

#pragma unroll 1
  for (int kb = 0; kb < S_LEN / 32 - 1; ++kb) {
    const int cur = kb & 1;
    const int key1 = (kb + 1) * 32;
    async_copy_b128(&ksm[1 - cur][tid * 8],
                    kbase + (size_t)key1 * HD + tid * 8);
    async_copy_b128(&vsm[1 - cur][srow * 32 + schk],
                    vbase + (size_t)srow * S_LEN + key1 + schk);
    wait_async2();                     // current buffer's 2 copies retired
    __syncthreads();
    flash_block(ksm[cur], vsm[cur], pstage[wib], aq0, aq1, o, rm, rs,
                nl, kc, mofs);
    __syncthreads();                   // reads done before buffer reuse
  }
  // peeled last key-block (buffer 1: (NB-1)&1 == 1)
  wait_async0();
  __syncthreads();
  flash_block(ksm[1], vsm[1], pstage[wib], aq0, aq1, o, rm, rs, nl, kc, mofs);

  // --- normalize, scatter to heads (m = s*2+b, col = h*64+d), absmax
  const int b = bh >> 4, h = bh & 15;
  float lmax = 0.0f;
#pragma unroll
  for (int t = 0; t < 4; ++t) {
#pragma unroll
    for (int r = 0; r < 8; ++r) {
      int m = r + mofs;
      int sg = qt * 16 + m;
      float v = o[t][r] / rs[r];
      lmax = fmaxf(lmax, fabsf(v));
      heads[(size_t)(sg * BATCH + b) * D_EMB + h * HD + t * 16 + nl] = v;
    }
  }
  lmax = wave_max32(lmax);
  if (lane == 0) atomicMax(slot, __float_as_uint(lmax));
}

// ---- output GEMM ----------------------------------------------------------
// y(4096x1024) = heads16 @ Wo_q^T + bo, f32 row-major out, absmax slot.
// Same branch-free LDS-staged B-panel scheme as gemm_xw_k.
__global__ void gemm_out_k(const half_t* __restrict__ Hm,
                           const half_t* __restrict__ Wo,
                           const float* __restrict__ bo,
                           float* __restrict__ y,
                           unsigned int* __restrict__ slot) {
  __shared__ alignas(16) half_t bsm[2][64 * 32];
  const int tid  = threadIdx.x;
  const int lane = tid & 31;
  const int wib  = tid >> 5;
  const int ng   = blockIdx.x % (D_EMB / 64);
  const int mtg  = blockIdx.x / (D_EMB / 64);
  const int mt   = mtg * 8 + wib;
  const int nl   = lane & 15;
  const int kc   = (lane >> 4) << 3;
  const int ngb  = ng * 64;
  const int srow = tid >> 2;
  const int soff = srow * 32 + (tid & 3) * 8;

  v8f acc[4] = {v8f{}, v8f{}, v8f{}, v8f{}};
  const half_t* xrow = Hm + (size_t)(mt * 16 + nl) * D_EMB + kc;
  const half_t* wst  = Wo + (size_t)(ngb + srow) * D_EMB + (tid & 3) * 8;

  async_copy_b128(&bsm[0][soff], wst);
  v16h a = load_ab16(xrow);

#pragma unroll 1
  for (int kk = 0; kk < D_EMB - 64; kk += 64) {
    async_copy_b128(&bsm[1][soff], wst + kk + 32);
    wait_async1();
    __syncthreads();
    __builtin_prefetch(xrow + kk + 512, 0, 3);
    v16h a1 = load_ab16(xrow + kk + 32);
    gemm_step(bsm[0], a, acc, nl, kc);
    __syncthreads();
    async_copy_b128(&bsm[0][soff], wst + kk + 64);
    wait_async1();
    __syncthreads();
    v16h a2 = load_ab16(xrow + kk + 64);
    gemm_step(bsm[1], a1, acc, nl, kc);
    a = a2;
    __syncthreads();
  }
  async_copy_b128(&bsm[1][soff], wst + (D_EMB - 32));
  wait_async1();
  __syncthreads();
  v16h a1 = load_ab16(xrow + (D_EMB - 32));
  gemm_step(bsm[0], a, acc, nl, kc);
  __syncthreads();
  wait_async0();
  __syncthreads();
  gemm_step(bsm[1], a1, acc, nl, kc);

  float lmax = 0.0f;
#pragma unroll
  for (int t = 0; t < 4; ++t) {
    int nglob = ngb + t * 16 + nl;
    float bv = bo[nglob];
#pragma unroll
    for (int r = 0; r < 8; ++r) {
      int m = mt * 16 + r + ((lane >> 4) << 3);
      float v = acc[t][r] + bv;
      lmax = fmaxf(lmax, fabsf(v));
      y[(size_t)m * D_EMB + nglob] = v;
    }
  }
  lmax = wave_max32(lmax);
  if (lane == 0) atomicMax(slot, __float_as_uint(lmax));
}

// ---------------------------------------------------------------------------

static inline int ew_grid(int n) { return (n + 255) / 256; }

extern "C" void kernel_launch(void* const* d_in, const int* in_sizes, int n_in,
                              void* d_out, int out_size, void* d_ws, size_t ws_size,
                              hipStream_t stream) {
  const float* query = (const float*)d_in[0];   // (2048, 2, 1024)
  const float* key   = (const float*)d_in[1];
  const float* value = (const float*)d_in[2];
  const float* Wi    = (const float*)d_in[3];   // (3072, 1024)
  const float* bi    = (const float*)d_in[4];   // (3072)
  const float* Wo    = (const float*)d_in[5];   // (1024, 1024)
  const float* bo    = (const float*)d_in[6];   // (1024)
  float* out = (float*)d_out;                   // (2048, 2, 1024) f32

  // workspace carve-out (256-B aligned bumps), ~84 MB total
  char* w = (char*)d_ws;
  auto carve = [&](size_t bytes) {
    char* p = w;
    w += (bytes + 255) & ~(size_t)255;
    return p;
  };
  unsigned int* slots = (unsigned int*)carve(64);
  // slot indices: 0 wi, 1 wo, 2 xq, 3 xk, 4 xv, 5 q/8, 6 heads, 7 y
  half_t* wi16   = (half_t*)carve((size_t)N3 * D_EMB * 2);
  half_t* wo16   = (half_t*)carve((size_t)D_EMB * D_EMB * 2);
  half_t* q16    = (half_t*)carve((size_t)MROWS * D_EMB * 2);
  half_t* k16    = (half_t*)carve((size_t)MROWS * D_EMB * 2);
  half_t* v16    = (half_t*)carve((size_t)MROWS * D_EMB * 2);
  half_t* qh     = (half_t*)carve((size_t)BH * S_LEN * HD * 2);
  half_t* kh     = (half_t*)carve((size_t)BH * S_LEN * HD * 2);
  half_t* vT     = (half_t*)carve((size_t)BH * HD * S_LEN * 2);
  float*  headsf = (float*) carve((size_t)MROWS * D_EMB * 4);
  half_t* h16    = (half_t*)carve((size_t)MROWS * D_EMB * 2);

  const int nWi = N3 * D_EMB, nWo = D_EMB * D_EMB, nX = MROWS * D_EMB;
  const int nQKV = BH * S_LEN * HD;

  init_scalars_k<<<1, 32, 0, stream>>>(slots);

  // weight fake_quant -> f16
  absmax_f32_k<<<1024, 256, 0, stream>>>(Wi, nWi, slots + 0);
  absmax_f32_k<<<512, 256, 0, stream>>>(Wo, nWo, slots + 1);
  fq_f32_to_f16_k<<<ew_grid(nWi), 256, 0, stream>>>(Wi, wi16, nWi, slots + 0);
  fq_f32_to_f16_k<<<ew_grid(nWo), 256, 0, stream>>>(Wo, wo16, nWo, slots + 1);

  // activations f32 -> f16 (unquantized, per reference)
  cvt_f32_f16_k<<<ew_grid(nX), 256, 0, stream>>>(query, q16, nX);
  cvt_f32_f16_k<<<ew_grid(nX), 256, 0, stream>>>(key, k16, nX);
  cvt_f32_f16_k<<<ew_grid(nX), 256, 0, stream>>>(value, v16, nX);

  // input projections: 32 row-groups x 48 B-panels = 1536 blocks x 8 waves
  gemm_xw_k<<<1536, 256, 0, stream>>>(q16, wi16, bi, qh, 0,        HD, 1, slots + 2);
  gemm_xw_k<<<1536, 256, 0, stream>>>(k16, wi16, bi, kh, D_EMB,    HD, 1, slots + 3);
  gemm_xw_k<<<1536, 256, 0, stream>>>(v16, wi16, bi, vT, 2*D_EMB,  1, S_LEN, slots + 4);

  // fake_quant K, V; Q gets fq, then fq(Q/8) with its own global scale
  fq_f16_inplace_k<<<ew_grid(nQKV), 256, 0, stream>>>(kh, nQKV, slots + 3, 1.0f, nullptr, 0.0f);
  fq_f16_inplace_k<<<ew_grid(nQKV), 256, 0, stream>>>(vT, nQKV, slots + 4, 1.0f, nullptr, 0.0f);
  fq_f16_inplace_k<<<ew_grid(nQKV), 256, 0, stream>>>(qh, nQKV, slots + 2, 1.0f, slots + 5, 0.125f);
  fq_f16_inplace_k<<<ew_grid(nQKV), 256, 0, stream>>>(qh, nQKV, slots + 5, 0.125f, nullptr, 0.0f);

  // flash attention: 32 heads * 128 q-tiles = 4096 waves, 8 waves/block
  flash_attn_k<<<512, 256, 0, stream>>>(qh, kh, vT, headsf, slots + 6);

  // heads fake_quant -> f16, then output projection (32 x 16 = 512 blocks)
  fq_f32_to_f16_k<<<ew_grid(nX), 256, 0, stream>>>(headsf, h16, nX, slots + 6);
  gemm_out_k<<<512, 256, 0, stream>>>(h16, wo16, bo, out, slots + 7);

  // final fake_quant in place on d_out
  fq_f32_inplace_k<<<ew_grid(nX), 256, 0, stream>>>(out, nX, slots + 7);
}
